// DefectAttractor_88304527606102
// MI455X (gfx1250) — matrix operand order repaired and checked by reading, not compiled, not run on hardware
//
#include <hip/hip_runtime.h>

typedef __attribute__((ext_vector_type(16))) _Float16 v16h;
typedef __attribute__((ext_vector_type(8)))  _Float16 v8h;
typedef __attribute__((ext_vector_type(4)))  _Float16 v4h;
typedef __attribute__((ext_vector_type(8)))  float    v8f;
typedef __attribute__((ext_vector_type(4)))  float    v4f;
typedef __attribute__((ext_vector_type(4)))  unsigned int u32x4;
typedef __attribute__((ext_vector_type(8)))  int      i32x8;
typedef __attribute__((ext_vector_type(4)))  int      i32x4;

#define DDIM        128
#define MSITES      1024
#define CHUNK_SITES 64                         // sites staged in LDS per step
#define NCHUNK      (MSITES / CHUNK_SITES)     // 16
#define ROW_HALFS   136                        // 128 data + 8 pad halfs (272 B)
#define CHUNK_DW    (CHUNK_SITES * 64u)        // 4096 dwords of payload
#define EPSF        1e-8f

__device__ __forceinline__ float dot4(v4f a, v4f b) {
    return a[0]*b[0] + a[1]*b[1] + a[2]*b[2] + a[3]*b[3];
}

__device__ __forceinline__ unsigned lds_off(const void* p) {
    // generic pointer to __shared__: low 32 bits are the LDS byte offset
    return (unsigned)(size_t)p;
}

#if __has_builtin(__builtin_amdgcn_tensor_load_to_lds)
#define HAVE_TDM 1
// TDM D#: 1D run of CHUNK_DW dwords, pad 4 dwords after every 64 dwords so
// each 256-B site row lands at a 272-B LDS stride (bank-skewed B reads).
__device__ __forceinline__ void tdm_issue(const _Float16* gsrc, unsigned dst_lds) {
    unsigned long long ga = (unsigned long long)(size_t)gsrc;
    u32x4 g0;
    g0[0] = 1u;                                              // count=1
    g0[1] = dst_lds;                                         // lds_addr
    g0[2] = (unsigned)ga;                                    // global_addr lo
    g0[3] = ((unsigned)(ga >> 32) & 0x01FFFFFFu) | 0x80000000u; // hi + type=2
    i32x8 g1;
    g1[0] = (int)((2u << 16)      // data_size = 4 B
                | (1u << 20)      // pad_enable
                | (5u << 22)      // pad_interval: 64 dwords
                | (3u << 25));    // pad_amount: 4 dwords
    g1[1] = (int)((CHUNK_DW & 0xFFFFu) << 16);               // tensor_dim0 lo
    g1[2] = (int)((CHUNK_DW >> 16) | (1u << 16));            // dim0 hi | dim1=1
    g1[3] = (int)((CHUNK_DW & 0xFFFFu) << 16);               // tile_dim0
    g1[4] = 1;                                               // tile_dim1=1
    g1[5] = (int)CHUNK_DW;                                   // dim0_stride lo
    g1[6] = 0;
    g1[7] = 0;
    i32x4 z4 = {0, 0, 0, 0};
#if defined(__clang_major__) && (__clang_major__ >= 23)
    i32x8 z8 = {0, 0, 0, 0, 0, 0, 0, 0};
    __builtin_amdgcn_tensor_load_to_lds(g0, g1, z4, z4, z8, 0);
#else
    __builtin_amdgcn_tensor_load_to_lds(g0, g1, z4, z4, 0);
#endif
}
#else
#define HAVE_TDM 0
#endif

// ---------------------------------------------------------------------------
// Prep: sites [M,128] f32 -> f16 into ws, plus ||s_j||^2 in f32.
// ---------------------------------------------------------------------------
__global__ void da_prep(const float* __restrict__ s,
                        _Float16* __restrict__ sh,
                        float* __restrict__ s2) {
    const int site = blockIdx.x;
    const int lane = threadIdx.x;               // 0..31
    const float* p = s + (size_t)site * DDIM + lane * 4;
    v4f v = *(const v4f*)p;
    v4h h = { (_Float16)v[0], (_Float16)v[1], (_Float16)v[2], (_Float16)v[3] };
    *(v4h*)(sh + (size_t)site * DDIM + lane * 4) = h;
    float ss = dot4(v, v);
#pragma unroll
    for (int m = 1; m < 32; m <<= 1) ss += __shfl_xor(ss, m, 32);
    if (lane == 0) s2[site] = ss;
}

// ---------------------------------------------------------------------------
// Main: wave = 16 rows of x vs all 1024 sites. Sites streamed to LDS by TDM
// (double-buffered); distances via v_wmma_f32_16x16x32_f16; exact f32 epilogue.
// ---------------------------------------------------------------------------
__global__ void __launch_bounds__(256) da_main(
    const float* __restrict__ x, const float* __restrict__ s32,
    const _Float16* __restrict__ sh, const float* __restrict__ s2,
    const float* __restrict__ rate_p, const float* __restrict__ coh_p,
    const float* __restrict__ phi_p, float* __restrict__ out)
{
    __shared__ __align__(16) _Float16 bsm[2][CHUNK_SITES * ROW_HALFS];
    __shared__ int nearest_sm[8][16];

    const int lane     = threadIdx.x & 31;
    const int wave     = threadIdx.x >> 5;
    const int row_base = (blockIdx.x * 8 + wave) * 16;
    const int m16      = lane & 15;
    const int hs       = lane >> 4;

    // ---- A fragments: 16 rows x 128 K of x, f32 -> f16 in-register ----
    v16h a[4];
    {
        const float* xrow = x + (size_t)(row_base + m16) * DDIM;
#pragma unroll
        for (int kk = 0; kk < 4; ++kk) {
            const int kb = kk * 32;
            const float* c0 = xrow + kb + 8 * hs;
            const float* c1 = xrow + kb + 16 + 8 * hs;
            v4f f0 = *(const v4f*)(c0);
            v4f f1 = *(const v4f*)(c0 + 4);
            v4f g0 = *(const v4f*)(c1);
            v4f g1 = *(const v4f*)(c1 + 4);
            v16h af;
#pragma unroll
            for (int i = 0; i < 4; ++i) {
                af[i]      = (_Float16)f0[i];
                af[4 + i]  = (_Float16)f1[i];
                af[8 + i]  = (_Float16)g0[i];
                af[12 + i] = (_Float16)g1[i];
            }
            a[kk] = af;
        }
    }

    float best[8];
    int   bidx[8];
#pragma unroll
    for (int r = 0; r < 8; ++r) { best[r] = 3.0e38f; bidx[r] = 0; }

#if HAVE_TDM
    if (wave == 0) tdm_issue(sh, lds_off(&bsm[0][0]));
#endif

    for (int c = 0; c < NCHUNK; ++c) {
        const int buf = c & 1;
#if HAVE_TDM
        if (wave == 0) __builtin_amdgcn_s_wait_tensorcnt(0);
        __syncthreads();                       // chunk c visible to all waves
        if (wave == 0 && c + 1 < NCHUNK)
            tdm_issue(sh + (size_t)(c + 1) * CHUNK_SITES * DDIM,
                      lds_off(&bsm[(c + 1) & 1][0]));
#else
        {   // cooperative fallback copy into the padded layout
            const _Float16* src = sh + (size_t)c * CHUNK_SITES * DDIM;
            for (int t = threadIdx.x; t < CHUNK_SITES * 16; t += 256) {
                const int rowi = t >> 4, ci = t & 15;
                *(v8h*)&bsm[buf][rowi * ROW_HALFS + ci * 8] =
                    *(const v8h*)(src + rowi * DDIM + ci * 8);
            }
            __syncthreads();
        }
#endif
        // B fragment: lane -> site (m16 within tile), K runs at 8*hs and 16+8*hs
        const _Float16* bbase = &bsm[buf][(size_t)m16 * ROW_HALFS + 8 * hs];
#pragma unroll
        for (int t = 0; t < CHUNK_SITES / 16; ++t) {
            const _Float16* bp = bbase + t * 16 * ROW_HALFS;
            v8f acc = {0.f, 0.f, 0.f, 0.f, 0.f, 0.f, 0.f, 0.f};
#pragma unroll
            for (int kk = 0; kk < 4; ++kk) {
                v8h b0 = *(const v8h*)(bp + kk * 32);
                v8h b1 = *(const v8h*)(bp + kk * 32 + 16);
                v16h b;
#pragma unroll
                for (int i = 0; i < 8; ++i) { b[i] = b0[i]; b[8 + i] = b1[i]; }
                acc = __builtin_amdgcn_wmma_f32_16x16x32_f16(
                    false, a[kk], false, b, (short)0, acc, false, false);
            }
            const int   col = c * CHUNK_SITES + t * 16 + m16;
            const float s2v = s2[col];
#pragma unroll
            for (int r = 0; r < 8; ++r) {
                const float score = s2v - 2.0f * acc[r];   // d2 - ||x||^2
                if (score < best[r]) { best[r] = score; bidx[r] = col; }
            }
        }
        __syncthreads();                       // done reading buf before reuse
    }

    // ---- per-row argmin across the 16 lanes of each half ----
#pragma unroll
    for (int r = 0; r < 8; ++r) {
        float v  = best[r];
        int   ix = bidx[r];
#pragma unroll
        for (int m = 1; m < 16; m <<= 1) {
            const float ov = __shfl_xor(v, m, 32);
            const int   oi = __shfl_xor(ix, m, 32);
            if (ov < v || (ov == v && oi < ix)) { v = ov; ix = oi; }
        }
        bidx[r] = ix;
    }

    // transpose nearest indices to LDS: slot r / half h  <->  row 8h + r
    if (m16 == 0) {
#pragma unroll
        for (int r = 0; r < 8; ++r) nearest_sm[wave][hs * 8 + r] = bidx[r];
    }
    __syncthreads();

    // ---- exact f32 epilogue: 8 lanes per row, 4 rows at a time ----
    const float rate = *rate_p;
    const float coh  = *coh_p;
    const float tphi = tanf(*phi_p);
    const int   rg   = lane >> 3;              // row sub-group 0..3
    const int   l8   = lane & 7;               // 16 features per lane

#pragma unroll
    for (int it = 0; it < 4; ++it) {
        const int rr  = it * 4 + rg;
        const int row = row_base + rr;
        const int j   = nearest_sm[wave][rr];
        const float* xr = x   + (size_t)row * DDIM + l8 * 16;
        const float* sr = s32 + (size_t)j   * DDIM + l8 * 16;
        float xx = 0.f, xd = 0.f, dd = 0.f;
        v4f dir[4];
#pragma unroll
        for (int q = 0; q < 4; ++q) {
            v4f xv = *(const v4f*)(xr + q * 4);
            v4f sv = *(const v4f*)(sr + q * 4);
            v4f dv = sv - xv;
            dir[q] = dv;
            xx += dot4(xv, xv);
            xd += dot4(xv, dv);
            dd += dot4(dv, dv);
        }
#pragma unroll
        for (int m = 1; m < 8; m <<= 1) {      // reduce within 8-lane group
            xx += __shfl_xor(xx, m, 32);
            xd += __shfl_xor(xd, m, 32);
            dd += __shfl_xor(dd, m, 32);
        }
        const float stress = rate * sqrtf(dd);
        const float normal = fabsf(rate * xd) / (sqrtf(xx) + EPSF);
        const float scale  = (stress > coh + normal * tphi ? 2.0f : 0.5f) * rate;
        float* orow = out + (size_t)row * DDIM + l8 * 16;
#pragma unroll
        for (int q = 0; q < 4; ++q)
            *(v4f*)(orow + q * 4) = dir[q] * scale;
    }
}

// ---------------------------------------------------------------------------
extern "C" void kernel_launch(void* const* d_in, const int* in_sizes, int n_in,
                              void* d_out, int out_size, void* d_ws, size_t ws_size,
                              hipStream_t stream) {
    const float* x    = (const float*)d_in[0];   // [N,128] f32
    const float* s    = (const float*)d_in[1];   // [M,128] f32
    const float* rate = (const float*)d_in[2];
    const float* coh  = (const float*)d_in[3];
    const float* phi  = (const float*)d_in[4];
    float* out = (float*)d_out;

    const int N = in_sizes[0] / DDIM;            // 131072

    _Float16* sh = (_Float16*)d_ws;                                    // 256 KB
    float*    s2 = (float*)((char*)d_ws +
                            (size_t)MSITES * DDIM * sizeof(_Float16)); // + 4 KB

    da_prep<<<MSITES, 32, 0, stream>>>(s, sh, s2);
    da_main<<<N / 128, 256, 0, stream>>>(x, s, sh, s2, rate, coh, phi, out);
}
